// Model_39676907882554
// MI455X (gfx1250) — matrix operand order repaired and checked by reading, not compiled, not run on hardware
//
#include <hip/hip_runtime.h>
#include <stdint.h>

// ---------------------------------------------------------------------------
// Flash-attention w/ explicit dropout mask for MI455X (gfx1250, wave32, WMMA)
// B=2 H=16 S=2048 D=64, fp32 in/out, bf16 WMMA operands, f32 accumulation.
//
// Per block (256 thr = 8 waves): one (b,h), 128 query rows (16 per wave).
// Loop over 64 chunks of 32 keys:
//   S^T tile = K_chunk * Q^T  (2 key-tiles x 2 K-dim steps = 4 WMMA)
//   online softmax (base-2), stats are per-lane scalars (query = lane%16)
//   P^T -> B-fragment via lane-pair shuffle + bf16 pack (no LDS round trip)
//   O^T += V^T * P^T          (4 d-tiles = 4 WMMA)
// ---------------------------------------------------------------------------

typedef __attribute__((ext_vector_type(16))) __bf16       v16bf;
typedef __attribute__((ext_vector_type(8)))  __bf16       v8bf;
typedef __attribute__((ext_vector_type(8)))  float        v8f;
typedef __attribute__((ext_vector_type(4)))  float        v4f;
typedef __attribute__((ext_vector_type(8)))  unsigned int v8u;

#define SEQ 2048
#define DD  64
#define KC  32
#define NCH (SEQ / KC)

static __device__ __forceinline__ unsigned int bfbits(float x) {
  return (unsigned int)__builtin_bit_cast(unsigned short, (__bf16)x);
}
static __device__ __forceinline__ unsigned int pack2(float lo, float hi2) {
  return bfbits(lo) | (bfbits(hi2) << 16);
}
static __device__ __forceinline__ v16bf cat8(v8bf a, v8bf b) {
  return __builtin_shufflevector(a, b, 0,1,2,3,4,5,6,7,8,9,10,11,12,13,14,15);
}

__global__ __launch_bounds__(256)
void fa_dropout_wmma(const float* __restrict__ Q, const float* __restrict__ K,
                     const float* __restrict__ V, const float* __restrict__ scale_p,
                     const float* __restrict__ Mask, float* __restrict__ Out)
{
  __shared__ alignas(16) __bf16 K_lds[KC][DD];     // keys x d       (4 KB)
  __shared__ alignas(16) __bf16 VT_lds[DD][40];    // d x keys, pad  (5 KB)

  const int bh   = blockIdx.y;
  const int lane = threadIdx.x & 31;
  const int wave = threadIdx.x >> 5;
  const int n16  = lane & 15;
  const int hi   = lane >> 4;                       // which 16-lane half
  const int q    = blockIdx.x * 128 + wave * 16 + n16;

  // fold softmax scale and log2(e): exp(x) == exp2(x*log2e)
  const float sc = scale_p[0] * 1.4426950408889634f;

  const float* qh   = Q    + (size_t)bh * SEQ * DD;
  const float* kh   = K    + (size_t)bh * SEQ * DD;
  const float* vh   = V    + (size_t)bh * SEQ * DD;
  const float* mrow = Mask + (size_t)bh * SEQ * SEQ + (size_t)q * SEQ;
  float*       oh   = Out  + (size_t)bh * SEQ * DD;

  // ---- Q as B-fragments (N = query = lane%16, K-dim = d) -------------------
  // B order used here (32xN bf16, contiguous, same as A): lanes 0-15 hold K-dim 0..15, lanes 16-31 hold 16..31
  v16bf qf[2];
  {
    const float* qp = qh + (size_t)q * DD;
    #pragma unroll
    for (int f = 0; f < 2; ++f) {
      const int d0 = f * 32 + hi * 16;
      #pragma unroll
      for (int j = 0; j < 16; ++j) qf[f][j] = (__bf16)(qp[d0 + j] * sc);
    }
  }

  v8f acc[4];
  #pragma unroll
  for (int dt = 0; dt < 4; ++dt) acc[dt] = (v8f)0.0f;
  float m = -3.0e38f, l = 0.0f;

  for (int kc = 0; kc < NCH; ++kc) {
    const int kbase = kc * KC;

    // ---- stage K (bf16) and V^T (bf16, padded rows) into LDS ---------------
    __syncthreads();
    {
      const int kk   = threadIdx.x >> 3;            // 0..31 chunk-local key
      const int dseg = (threadIdx.x & 7) * 8;       // 0..56 step 8
      const float* kp = kh + (size_t)(kbase + kk) * DD + dseg;
      const float* vp = vh + (size_t)(kbase + kk) * DD + dseg;
      v4f k0 = *(const v4f*)kp,     k1 = *(const v4f*)(kp + 4);
      v4f v0 = *(const v4f*)vp,     v1 = *(const v4f*)(vp + 4);
      v8bf kb;
      #pragma unroll
      for (int i = 0; i < 4; ++i) { kb[i] = (__bf16)k0[i]; kb[4+i] = (__bf16)k1[i]; }
      *(v8bf*)&K_lds[kk][dseg] = kb;
      #pragma unroll
      for (int i = 0; i < 4; ++i) {
        VT_lds[dseg + i][kk]     = (__bf16)v0[i];
        VT_lds[dseg + 4 + i][kk] = (__bf16)v1[i];
      }
    }
    __syncthreads();

    // ---- dropout mask for this lane's P^T elements (issue early) -----------
    const float* mp = mrow + kbase + hi * 8;
    v4f m0a = *(const v4f*)(mp);          v4f m0b = *(const v4f*)(mp + 4);
    v4f m1a = *(const v4f*)(mp + 16);     v4f m1b = *(const v4f*)(mp + 20);
    if (kc + 1 < NCH) __builtin_prefetch(mrow + kbase + KC, 0, 1);

    // ---- S^T = K_chunk * Q^T  (two 16-key tiles, chained over d) ----------
    v8f c0 = (v8f)0.0f, c1 = (v8f)0.0f;
    #pragma unroll
    for (int f = 0; f < 2; ++f) {
      const int dlo = f * 32 + hi * 16;             // A layout: K-dim contiguous
      const int dhi2 = f * 32 + hi * 16 + 8;
      v16bf a0 = cat8(*(const v8bf*)&K_lds[n16][dlo],
                      *(const v8bf*)&K_lds[n16][dhi2]);
      c0 = __builtin_amdgcn_wmma_f32_16x16x32_bf16(false, a0, false, qf[f],
                                                   (short)0, c0, false, false);
      v16bf a1 = cat8(*(const v8bf*)&K_lds[16 + n16][dlo],
                      *(const v8bf*)&K_lds[16 + n16][dhi2]);
      c1 = __builtin_amdgcn_wmma_f32_16x16x32_bf16(false, a1, false, qf[f],
                                                   (short)0, c1, false, false);
    }

    // ---- online softmax (query row = 2 lanes -> single xor-16 reduce) ------
    float cm = c0[0];
    #pragma unroll
    for (int r = 1; r < 8; ++r) cm = fmaxf(cm, c0[r]);
    #pragma unroll
    for (int r = 0; r < 8; ++r) cm = fmaxf(cm, c1[r]);
    cm = fmaxf(cm, __shfl_xor(cm, 16));
    const float mn = fmaxf(m, cm);
    const float al = __builtin_amdgcn_exp2f(m - mn);

    float p0[8], p1[8], ps = 0.0f;
    #pragma unroll
    for (int r = 0; r < 8; ++r) {
      p0[r] = __builtin_amdgcn_exp2f(c0[r] - mn); ps += p0[r];
      p1[r] = __builtin_amdgcn_exp2f(c1[r] - mn); ps += p1[r];
    }
    ps += __shfl_xor(ps, 16);
    l = l * al + ps;                 // denominator excludes dropout mask
    m = mn;
    #pragma unroll
    for (int dt = 0; dt < 4; ++dt) acc[dt] *= al;

    // ---- apply mask (1/(1-p) folded into epilogue), pack P^T B-fragment ----
    #pragma unroll
    for (int i = 0; i < 4; ++i) {
      p0[i] *= m0a[i]; p0[4+i] *= m0b[i];
      p1[i] *= m1a[i]; p1[4+i] *= m1b[i];
    }
    unsigned int x0[4], x1[4];
    #pragma unroll
    for (int j = 0; j < 4; ++j) { x0[j] = pack2(p0[2*j], p0[2*j+1]);
                                  x1[j] = pack2(p1[2*j], p1[2*j+1]); }
    v8u w;
    #pragma unroll
    for (int j = 0; j < 4; ++j) {
      unsigned int y0 = (unsigned int)__shfl_xor((int)x0[j], 16); // partner keys
      unsigned int y1 = (unsigned int)__shfl_xor((int)x1[j], 16);
      w[j]     = hi ? y1    : x0[j];   // halves 0-7 : keys {0..7 | 16..23}
      w[4 + j] = hi ? x1[j] : y0;      // halves 8-15: keys {8..15| 24..31}
    }
    v16bf pf = __builtin_bit_cast(v16bf, w);

    // ---- O^T += V^T * P^T --------------------------------------------------
    #pragma unroll
    for (int dt = 0; dt < 4; ++dt) {
      v16bf va = cat8(*(const v8bf*)&VT_lds[dt*16 + n16][hi * 16],
                      *(const v8bf*)&VT_lds[dt*16 + n16][hi * 16 + 8]);
      acc[dt] = __builtin_amdgcn_wmma_f32_16x16x32_bf16(false, va, false, pf,
                                                        (short)0, acc[dt],
                                                        false, false);
    }
  }

  // ---- epilogue: out = O^T * (mask_scale 2.0 / rowsum), scatter transposed -
  const float inv = 2.0f / l;
  #pragma unroll
  for (int dt = 0; dt < 4; ++dt) {
    #pragma unroll
    for (int r = 0; r < 8; ++r) {
      const int d = dt * 16 + hi * 8 + r;
      oh[(size_t)q * DD + d] = acc[dt][r] * inv;
    }
  }
}

extern "C" void kernel_launch(void* const* d_in, const int* in_sizes, int n_in,
                              void* d_out, int out_size, void* d_ws, size_t ws_size,
                              hipStream_t stream) {
  (void)in_sizes; (void)n_in; (void)out_size; (void)d_ws; (void)ws_size;
  const float* Qp = (const float*)d_in[0];
  const float* Kp = (const float*)d_in[1];
  const float* Vp = (const float*)d_in[2];
  const float* Sp = (const float*)d_in[3];   // scale_factor (1 element)
  const float* Mp = (const float*)d_in[4];   // dropout keep-mask
  float* Op = (float*)d_out;

  dim3 grid(SEQ / 128, 2 * 16);              // 16 q-tiles x (B*H)
  fa_dropout_wmma<<<grid, 256, 0, stream>>>(Qp, Kp, Vp, Sp, Mp, Op);
}
